// Attention_28269474742408
// MI455X (gfx1250) — matrix operand order repaired and checked by reading, not compiled, not run on hardware
//
#include <hip/hip_runtime.h>
#include <hip/hip_bf16.h>

// ---------------------------------------------------------------------------
// MHA block: q,k,v = x@W{q,k,v}^T + b ; softmax(q k^T / sqrt(D)) v ; LN ; @Wo^T+bo
// B=8 N=1024 C=768 H=12 D=64.  All matmuls via v_wmma_f32_16x16x32_f16.
// ---------------------------------------------------------------------------

typedef __attribute__((ext_vector_type(16))) _Float16 v16h;
typedef __attribute__((ext_vector_type(8)))  _Float16 v8h;
typedef __attribute__((ext_vector_type(4)))  _Float16 v4h;
typedef __attribute__((ext_vector_type(8)))  float    v8f;

#define BB 8
#define NN 1024
#define CC 768
#define HH 12
#define DD 64
#define MM (BB * NN)   // 8192 rows

__device__ __forceinline__ v8f wmma32(v16h a, v16h b, v8f c) {
  // D = A(16x32 f16) * B(32x16 f16) + C(16x16 f32)
  return __builtin_amdgcn_wmma_f32_16x16x32_f16(
      /*neg_a=*/false, a, /*neg_b=*/false, b,
      /*c_mod=*/(short)0, c, /*reuse_a=*/false, /*reuse_b=*/false);
}

// A fragment: 16x32 f16 tile, rows m0..m0+15, k0..k0+31, from row-major [*, ld].
// ISA layout: lanes 0-15 row=lane, VGPR0-3 K=0..7, VGPR4-7 K=16..23;
//             lanes 16-31 same rows, K=8..15 and K=24..31.
__device__ __forceinline__ v16h load_a_frag(const _Float16* base, int ld, int m0, int k0) {
  const int lane = threadIdx.x & 31;
  const int hf   = lane >> 4;
  const int row  = lane & 15;
  const _Float16* p = base + (size_t)(m0 + row) * ld + k0 + 8 * hf;
  v8h lo = *(const v8h*)(p);
  v8h hi = *(const v8h*)(p + 16);
  v16h r;
#pragma unroll
  for (int i = 0; i < 8; ++i) { r[i] = lo[i]; r[i + 8] = hi[i]; }
  return r;
}

// B fragment: 32x16 (KxN) where B[k][n] = S[n0+n][k0+k] with S row-major [*, ld]
// (i.e. S plays the role of the transposed B: weights W[n,k], or K / V^T tiles).
// ISA layout: lane n=lane&15; lanes 0-15 hold K=0..7,16..23; lanes 16-31 K=8..15,24..31.
__device__ __forceinline__ v16h load_bT_frag(const _Float16* base, int ld, int n0, int k0) {
  const int lane = threadIdx.x & 31;
  const int hk   = lane >> 4;
  const int col  = lane & 15;
  const _Float16* p = base + (size_t)(n0 + col) * ld + k0 + 8 * hk;
  v8h lo = *(const v8h*)(p);
  v8h hi = *(const v8h*)(p + 16);
  v16h r;
#pragma unroll
  for (int i = 0; i < 8; ++i) { r[i] = lo[i]; r[i + 8] = hi[i]; }
  return r;
}

// ---------------------------------------------------------------------------
// Kernel 1: fp32 -> fp16 conversion (x and the four weight matrices)
// ---------------------------------------------------------------------------
__global__ void cvt_f16_kernel(const float* __restrict__ s, _Float16* __restrict__ d, int n4) {
  int i = blockIdx.x * blockDim.x + threadIdx.x;
  if (i < n4) {
    float4 v = ((const float4*)s)[i];
    v4h o;
    o[0] = (_Float16)v.x; o[1] = (_Float16)v.y;
    o[2] = (_Float16)v.z; o[3] = (_Float16)v.w;
    ((v4h*)d)[i] = o;
  }
}

// ---------------------------------------------------------------------------
// Kernel 2: fused QKV projection.  grid=(MM/128, CC/64, 3), block=128 (4 waves).
// Each wave computes a 32x64 output tile (2 A-frags x 4 B-frags -> 8 WMMA/k-step,
// each B fragment reused twice).
// z=0 -> Q (scaled by D^-0.5, layout [B,H,N,D] f16)
// z=1 -> K (layout [B,H,N,D] f16)
// z=2 -> V stored transposed (layout [B,H,D,N] f16) so P@V B-frags are contiguous
// ---------------------------------------------------------------------------
__global__ void qkv_gemm_kernel(const _Float16* __restrict__ X,
                                const _Float16* __restrict__ Wq,
                                const _Float16* __restrict__ Wk,
                                const _Float16* __restrict__ Wv,
                                const float* __restrict__ bq,
                                const float* __restrict__ bk,
                                const float* __restrict__ bv,
                                _Float16* __restrict__ Q,
                                _Float16* __restrict__ K,
                                _Float16* __restrict__ VT) {
  const int wave = threadIdx.x >> 5;
  const int lane = threadIdx.x & 31;
  const int m0 = blockIdx.x * 128 + wave * 32;
  const int n0 = blockIdx.y * 64;
  const int which = blockIdx.z;
  const _Float16* W  = (which == 0) ? Wq : (which == 1) ? Wk : Wv;
  const float*    bi = (which == 0) ? bq : (which == 1) ? bk : bv;

  v8f acc[2][4] = {};
  for (int k0 = 0; k0 < CC; k0 += 32) {
    v16h a0 = load_a_frag(X, CC, m0, k0);
    v16h a1 = load_a_frag(X, CC, m0 + 16, k0);
#pragma unroll
    for (int nt = 0; nt < 4; ++nt) {
      v16h bf = load_bT_frag(W, CC, n0 + nt * 16, k0);
      acc[0][nt] = wmma32(a0, bf, acc[0][nt]);
      acc[1][nt] = wmma32(a1, bf, acc[1][nt]);
    }
  }

  const int hf  = lane >> 4;
  const int col = lane & 15;
#pragma unroll
  for (int g = 0; g < 2; ++g) {
#pragma unroll
    for (int nt = 0; nt < 4; ++nt) {
      const int c  = n0 + nt * 16 + col;     // global output column 0..767
      const int h  = c >> 6;                 // head
      const int d  = c & 63;                 // dim within head
      const float bv_ = bi[c];
#pragma unroll
      for (int r = 0; r < 8; ++r) {
        const int m  = m0 + g * 16 + r + 8 * hf;   // global row 0..8191
        const int b  = m >> 10;
        const int ns = m & (NN - 1);
        const float v = acc[g][nt][r] + bv_;
        if (which == 0) {
          Q[(((size_t)(b * HH + h)) * NN + ns) * DD + d] = (_Float16)(v * 0.125f);
        } else if (which == 1) {
          K[(((size_t)(b * HH + h)) * NN + ns) * DD + d] = (_Float16)v;
        } else {
          VT[(((size_t)(b * HH + h)) * DD + d) * NN + ns] = (_Float16)v;
        }
      }
    }
  }
}

// ---------------------------------------------------------------------------
// Kernel 3: attention.  grid = B*H*(N/64) = 1536 blocks of 128 threads.
// Each wave owns 16 query rows. Pass 1: exact row maxima via score WMMAs +
// 16-lane butterfly. Pass 2: exp in fp32, transpose P tile through per-wave LDS
// (C-fragment layout -> A-fragment layout), accumulate P@V with WMMA.
// Pbuf is strictly per-wave: ordering needs only a workgroup-scope fence
// (s_wait_dscnt), not s_barrier; DS ops of one wave execute in order (ISA 7.1).
// Output fp32 into attn buffer laid out [B,N,C].
// ---------------------------------------------------------------------------
__global__ void attn_kernel(const _Float16* __restrict__ Q,
                            const _Float16* __restrict__ K,
                            const _Float16* __restrict__ VT,
                            float* __restrict__ O) {
  __shared__ __align__(16) _Float16 Pbuf[4][16][32];   // per-wave 16x32 P tile

  const int wave = threadIdx.x >> 5;
  const int lane = threadIdx.x & 31;
  const int hf   = lane >> 4;
  const int qblk = blockIdx.x & 15;        // 16 blocks of 64 query rows
  const int bh   = blockIdx.x >> 4;        // 0..95
  const int qbase = qblk * 64 + wave * 16;

  const _Float16* qh = Q  + (size_t)bh * NN * DD;
  const _Float16* kh = K  + (size_t)bh * NN * DD;
  const _Float16* vt = VT + (size_t)bh * DD * NN;

  const v16h qa0 = load_a_frag(qh, DD, qbase, 0);
  const v16h qa1 = load_a_frag(qh, DD, qbase, 32);

  // ---- pass 1: row maxima over all 1024 keys ----
  float rmax[8];
#pragma unroll
  for (int r = 0; r < 8; ++r) rmax[r] = -1e30f;
  for (int j0 = 0; j0 < NN; j0 += 16) {
    v16h kb0 = load_bT_frag(kh, DD, j0, 0);
    v16h kb1 = load_bT_frag(kh, DD, j0, 32);
    v8f s = {};
    s = wmma32(qa0, kb0, s);
    s = wmma32(qa1, kb1, s);
#pragma unroll
    for (int r = 0; r < 8; ++r) rmax[r] = fmaxf(rmax[r], s[r]);
  }
#pragma unroll
  for (int m = 1; m < 16; m <<= 1) {
#pragma unroll
    for (int r = 0; r < 8; ++r) rmax[r] = fmaxf(rmax[r], __shfl_xor(rmax[r], m, 32));
  }

  // ---- pass 2: exp, P@V ----
  float rsum[8] = {0.f, 0.f, 0.f, 0.f, 0.f, 0.f, 0.f, 0.f};
  v8f oacc[4] = {};
  for (int j0 = 0; j0 < NN; j0 += 32) {
    v8f s0 = {}, s1 = {};
    {
      v16h kb0 = load_bT_frag(kh, DD, j0, 0);
      v16h kb1 = load_bT_frag(kh, DD, j0, 32);
      s0 = wmma32(qa0, kb0, s0);
      s0 = wmma32(qa1, kb1, s0);
      kb0 = load_bT_frag(kh, DD, j0 + 16, 0);
      kb1 = load_bT_frag(kh, DD, j0 + 16, 32);
      s1 = wmma32(qa0, kb0, s1);
      s1 = wmma32(qa1, kb1, s1);
    }
#pragma unroll
    for (int r = 0; r < 8; ++r) {
      const float p0 = __expf(s0[r] - rmax[r]);
      const float p1 = __expf(s1[r] - rmax[r]);
      rsum[r] += p0 + p1;
      Pbuf[wave][r + 8 * hf][lane & 15]        = (_Float16)p0;
      Pbuf[wave][r + 8 * hf][(lane & 15) + 16] = (_Float16)p1;
    }
    // Order per-wave LDS stores before the cross-lane LDS reads (s_wait_dscnt,
    // no s_barrier needed: Pbuf regions are private to each wave and same-wave
    // DS instructions execute in order; WAR on the next iteration's stores is
    // likewise covered by in-order DS).
    __builtin_amdgcn_fence(__ATOMIC_ACQ_REL, "workgroup");
    v16h pa = load_a_frag(&Pbuf[wave][0][0], 32, 0, 0);
#pragma unroll
    for (int dt = 0; dt < 4; ++dt) {
      v16h vb = load_bT_frag(vt, NN, dt * 16, j0);
      oacc[dt] = wmma32(pa, vb, oacc[dt]);
    }
  }

#pragma unroll
  for (int m = 1; m < 16; m <<= 1) {
#pragma unroll
    for (int r = 0; r < 8; ++r) rsum[r] += __shfl_xor(rsum[r], m, 32);
  }

  const int b = bh / HH, h = bh % HH;
#pragma unroll
  for (int r = 0; r < 8; ++r) {
    const float inv = 1.0f / rsum[r];
    const int row = qbase + r + 8 * hf;
    float* op = O + ((size_t)(b * NN + row)) * CC + h * DD;
#pragma unroll
    for (int dt = 0; dt < 4; ++dt)
      op[dt * 16 + (lane & 15)] = oacc[dt][r] * inv;
  }
}

// ---------------------------------------------------------------------------
// Kernel 4: LayerNorm over C=768 per row, fp32 in -> fp16 out.  grid=8192, blk=256.
// ---------------------------------------------------------------------------
__global__ void ln_kernel(const float* __restrict__ A,
                          const float* __restrict__ g,
                          const float* __restrict__ bta,
                          _Float16* __restrict__ Hf) {
  __shared__ float ssum[8], ssq[8];
  const int row = blockIdx.x;
  const float* a = A + (size_t)row * CC;
  const int t = threadIdx.x;
  const float x0 = a[t], x1 = a[t + 256], x2 = a[t + 512];
  float s = x0 + x1 + x2;
  float q = x0 * x0 + x1 * x1 + x2 * x2;
#pragma unroll
  for (int m = 1; m < 32; m <<= 1) {
    s += __shfl_xor(s, m, 32);
    q += __shfl_xor(q, m, 32);
  }
  const int wave = t >> 5, lane = t & 31;
  if (lane == 0) { ssum[wave] = s; ssq[wave] = q; }
  __syncthreads();
  float ts = 0.f, tq = 0.f;
#pragma unroll
  for (int i = 0; i < 8; ++i) { ts += ssum[i]; tq += ssq[i]; }
  const float mu   = ts * (1.0f / (float)CC);
  const float var  = tq * (1.0f / (float)CC) - mu * mu;
  const float rstd = rsqrtf(var + 1e-5f);
  _Float16* hp = Hf + (size_t)row * CC;
  int c = t;
  hp[c] = (_Float16)((x0 - mu) * rstd * g[c] + bta[c]); c += 256;
  hp[c] = (_Float16)((x1 - mu) * rstd * g[c] + bta[c]); c += 256;
  hp[c] = (_Float16)((x2 - mu) * rstd * g[c] + bta[c]);
}

// ---------------------------------------------------------------------------
// Kernel 5: output projection: out = Hf @ Wo^T + bo, fp32 out.
// grid=(MM/128, CC/64), block=128; 32x64 tile per wave like kernel 2.
// ---------------------------------------------------------------------------
__global__ void out_gemm_kernel(const _Float16* __restrict__ Hf,
                                const _Float16* __restrict__ Wo,
                                const float* __restrict__ bo,
                                float* __restrict__ Out) {
  const int wave = threadIdx.x >> 5;
  const int lane = threadIdx.x & 31;
  const int m0 = blockIdx.x * 128 + wave * 32;
  const int n0 = blockIdx.y * 64;

  v8f acc[2][4] = {};
  for (int k0 = 0; k0 < CC; k0 += 32) {
    v16h a0 = load_a_frag(Hf, CC, m0, k0);
    v16h a1 = load_a_frag(Hf, CC, m0 + 16, k0);
#pragma unroll
    for (int nt = 0; nt < 4; ++nt) {
      v16h bf = load_bT_frag(Wo, CC, n0 + nt * 16, k0);
      acc[0][nt] = wmma32(a0, bf, acc[0][nt]);
      acc[1][nt] = wmma32(a1, bf, acc[1][nt]);
    }
  }
  const int hf = lane >> 4, col = lane & 15;
#pragma unroll
  for (int g = 0; g < 2; ++g) {
#pragma unroll
    for (int nt = 0; nt < 4; ++nt) {
      const int c = n0 + nt * 16 + col;
      const float bv = bo[c];
#pragma unroll
      for (int r = 0; r < 8; ++r)
        Out[(size_t)(m0 + g * 16 + r + 8 * hf) * CC + c] = acc[g][nt][r] + bv;
    }
  }
}

// ---------------------------------------------------------------------------
extern "C" void kernel_launch(void* const* d_in, const int* in_sizes, int n_in,
                              void* d_out, int out_size, void* d_ws, size_t ws_size,
                              hipStream_t stream) {
  const float* x    = (const float*)d_in[0];
  const float* Wq   = (const float*)d_in[1];
  const float* bq   = (const float*)d_in[2];
  const float* Wk   = (const float*)d_in[3];
  const float* bk   = (const float*)d_in[4];
  const float* Wv   = (const float*)d_in[5];
  const float* bv   = (const float*)d_in[6];
  const float* Wo   = (const float*)d_in[7];
  const float* bo   = (const float*)d_in[8];
  const float* ln_g = (const float*)d_in[9];
  const float* ln_b = (const float*)d_in[10];
  float* out = (float*)d_out;

  // workspace carve-up (all 16B-aligned region sizes)
  char* ws = (char*)d_ws;
  _Float16* x_h  = (_Float16*)ws; ws += (size_t)MM * CC * 2;   // 12.6 MB
  _Float16* Wq_h = (_Float16*)ws; ws += (size_t)CC * CC * 2;
  _Float16* Wk_h = (_Float16*)ws; ws += (size_t)CC * CC * 2;
  _Float16* Wv_h = (_Float16*)ws; ws += (size_t)CC * CC * 2;
  _Float16* Wo_h = (_Float16*)ws; ws += (size_t)CC * CC * 2;
  _Float16* Qb   = (_Float16*)ws; ws += (size_t)MM * CC * 2;   // [B,H,N,D]
  _Float16* Kb   = (_Float16*)ws; ws += (size_t)MM * CC * 2;   // [B,H,N,D]
  _Float16* VTb  = (_Float16*)ws; ws += (size_t)MM * CC * 2;   // [B,H,D,N]
  float*    Ab   = (float*)ws;    ws += (size_t)MM * CC * 4;   // attn fp32 [B,N,C]
  _Float16* Hfb  = (_Float16*)ws; ws += (size_t)MM * CC * 2;   // post-LN f16

  // 1) fp32 -> fp16 conversions
  {
    const int n4x = MM * CC / 4;          // 1,572,864
    const int n4w = CC * CC / 4;          // 147,456
    cvt_f16_kernel<<<(n4x + 255) / 256, 256, 0, stream>>>(x,  x_h,  n4x);
    cvt_f16_kernel<<<(n4w + 255) / 256, 256, 0, stream>>>(Wq, Wq_h, n4w);
    cvt_f16_kernel<<<(n4w + 255) / 256, 256, 0, stream>>>(Wk, Wk_h, n4w);
    cvt_f16_kernel<<<(n4w + 255) / 256, 256, 0, stream>>>(Wv, Wv_h, n4w);
    cvt_f16_kernel<<<(n4w + 255) / 256, 256, 0, stream>>>(Wo, Wo_h, n4w);
  }

  // 2) QKV projections (WMMA, 32x64 tile per wave)
  qkv_gemm_kernel<<<dim3(MM / 128, CC / 64, 3), 128, 0, stream>>>(
      x_h, Wq_h, Wk_h, Wv_h, bq, bk, bv, Qb, Kb, VTb);

  // 3) attention (WMMA scores + WMMA P@V)
  attn_kernel<<<BB * HH * (NN / 64), 128, 0, stream>>>(Qb, Kb, VTb, Ab);

  // 4) layernorm
  ln_kernel<<<MM, 256, 0, stream>>>(Ab, ln_g, ln_b, Hfb);

  // 5) output projection (WMMA) -> fp32 d_out
  out_gemm_kernel<<<dim3(MM / 128, CC / 64), 128, 0, stream>>>(Hfb, Wo_h, bo, out);

  (void)in_sizes; (void)n_in; (void)out_size; (void)ws_size;
}
